// _RefVisionAttention_25159918420624
// MI455X (gfx1250) — matrix operand order repaired and checked by reading, not compiled, not run on hardware
//
#include <hip/hip_runtime.h>
#include <hip/hip_bf16.h>

// ---------------- problem constants ----------------
#define B_    4
#define S_    2048
#define HID_  1536
#define NH_   16
#define NKV_  8
#define HD_   96
#define QKVN_ 3072        // NH*HD + 2*NKV*HD
#define NEGINF_ (-3.4028234663852886e38f)

typedef unsigned short u16;
typedef __attribute__((ext_vector_type(8)))  unsigned short u16x8;
typedef __attribute__((ext_vector_type(16))) __bf16        v16bf;
typedef __attribute__((ext_vector_type(8)))  float         v8f;

union Frag16 { v16bf bf; u16x8 h[2]; unsigned short u[16]; };

static __device__ __forceinline__ u16 f2bf(float f) {
    unsigned u = __float_as_uint(f);
    u += 0x7fffu + ((u >> 16) & 1u);      // round-to-nearest-even
    return (u16)(u >> 16);
}

// LDS byte offset of a generic pointer to __shared__ (generic LDS addr low 32 bits = LDS offset)
static __device__ __forceinline__ unsigned lds_off(const void* p) {
    return (unsigned)(unsigned long long)p;
}

// async global->LDS copies (CDNA5 VGLOBAL opcodes 97/98, tracked by ASYNCcnt)
static __device__ __forceinline__ void async_b128(unsigned lds, const void* gaddr) {
    asm volatile("global_load_async_to_lds_b128 %0, %1, off" :: "v"(lds), "v"(gaddr) : "memory");
}
static __device__ __forceinline__ void async_b64(unsigned lds, const void* gaddr) {
    asm volatile("global_load_async_to_lds_b64 %0, %1, off" :: "v"(lds), "v"(gaddr) : "memory");
}
#if defined(__has_builtin) && __has_builtin(__builtin_amdgcn_s_wait_asynccnt)
#define WAIT_ASYNC(n) __builtin_amdgcn_s_wait_asynccnt(n)
#else
#define WAIT_ASYNC(n) asm volatile("s_wait_asynccnt %0" :: "i"(n) : "memory")
#endif

// ---------------- kernel 0: fp32 -> bf16 convert ----------------
__global__ void cvt_f32_bf16(const float* __restrict__ s, u16* __restrict__ d, int n) {
    int i = blockIdx.x * blockDim.x + threadIdx.x;
    if (i < n) d[i] = f2bf(s[i]);
}

// ---------------- kernel 1/6: C = A(MxK) * Bsrc(NxK)^T, bf16 WMMA ----------------
// 64x64 block tile, 256 threads (8 waves), wave = 16x32 tile, K-step 32.
// Double-buffered LDS filled by async global->LDS DMA (2 async instrs per wave per tile).
__global__ __launch_bounds__(256) void gemm_nt_bf16(
    const u16* __restrict__ A, const u16* __restrict__ Bsrc,
    float* __restrict__ C, int M, int N, int K)
{
    __shared__ __align__(16) u16 As[2][64][40];
    __shared__ __align__(16) u16 Bs[2][64][40];
    const int m0 = blockIdx.y * 64, n0 = blockIdx.x * 64;
    const int t = threadIdx.x, lane = t & 31, wid = t >> 5;
    const int wrow = (wid & 3) * 16, wcol = (wid >> 2) * 32;
    const int lr = lane & 15, hi = lane >> 4;
    const int row = t >> 2, seg = (t & 3) * 8;
    v8f acc0 = {}, acc1 = {};

    auto issue = [&](int k0, int buf) {
        async_b128(lds_off(&As[buf][row][seg]), &A   [(size_t)(m0 + row) * K + k0 + seg]);
        async_b128(lds_off(&Bs[buf][row][seg]), &Bsrc[(size_t)(n0 + row) * K + k0 + seg]);
    };

    const int nt = K >> 5;
    issue(0, 0);
    for (int i = 0; i < nt; ++i) {
        const int buf = i & 1;
        if (i + 1 < nt) { issue((i + 1) << 5, buf ^ 1); WAIT_ASYNC(2); }
        else            { WAIT_ASYNC(0); }
        __syncthreads();

        Frag16 a, b0, b1;
        const u16* ap = &As[buf][wrow + lr][0];
        a.h[0] = *(const u16x8*)&ap[hi * 8];          // K 0-7 / 8-15
        a.h[1] = *(const u16x8*)&ap[16 + hi * 8];     // K 16-23 / 24-31
        const u16* bp0 = &Bs[buf][wcol + lr][hi * 8];
        b0.h[0] = *(const u16x8*)&bp0[0];
        b0.h[1] = *(const u16x8*)&bp0[16];
        const u16* bp1 = &Bs[buf][wcol + 16 + lr][hi * 8];
        b1.h[0] = *(const u16x8*)&bp1[0];
        b1.h[1] = *(const u16x8*)&bp1[16];

        acc0 = __builtin_amdgcn_wmma_f32_16x16x32_bf16(false, a.bf, false, b0.bf, (short)0, acc0, false, false);
        acc1 = __builtin_amdgcn_wmma_f32_16x16x32_bf16(false, a.bf, false, b1.bf, (short)0, acc1, false, false);
        __syncthreads();
    }
#pragma unroll
    for (int r = 0; r < 8; ++r) {
        const int m = m0 + wrow + r + hi * 8;
        float* cp = &C[(size_t)m * N + n0 + wcol + lr];
        cp[0]  = acc0[r];
        cp[16] = acc1[r];
    }
}

// ---------------- kernel 2: RMSNorm + mRoPE, write bf16 Q/K/Vt ----------------
// blockDim = 96 (one channel per thread), one (b,s,head-row) per block.
// rows per (b,s): hh 0-15 = q heads, 16-23 = k heads, 24-31 = v heads.
__global__ void normrope_kernel(
    const float* __restrict__ qkv, const float* __restrict__ cosp,
    const float* __restrict__ sinp, const float* __restrict__ qw,
    const float* __restrict__ kw,
    u16* __restrict__ Qb, u16* __restrict__ Kb, u16* __restrict__ Vt)
{
    const int rid = blockIdx.x;
    const int hh = rid & 31;
    const int bs = rid >> 5;              // b*S + s
    const int b  = bs >> 11;              // S = 2048
    const int s  = bs & (S_ - 1);
    const int c  = threadIdx.x;           // 0..95

    __shared__ float sq[96];
    __shared__ float xb[96];
    __shared__ float rmss;

    const float x = qkv[(size_t)bs * QKVN_ + hh * HD_ + c];
    sq[c] = x * x;
    __syncthreads();
    if (c < 32) {
        float v = sq[c] + sq[c + 32] + sq[c + 64];
#pragma unroll
        for (int off = 16; off; off >>= 1) v += __shfl_xor(v, off, 32);
        if (c == 0) rmss = rsqrtf(v * (1.0f / HD_) + 1e-6f);
    }
    __syncthreads();
    const float w  = (hh < 16) ? qw[c] : (hh < 24 ? kw[c] : 1.0f);
    const float xn = x * rmss * w;
    xb[c] = xn;
    __syncthreads();

    float outv;
    if (hh < 24) {  // q and k get mRoPE (NDIM=2, NR=48, rotate_half +-24)
        const float cs = cosp[(size_t)bs * HD_ + c];
        const float sn = sinp[(size_t)bs * HD_ + c];
        const int j = c - (c / 48) * 48;
        const float rot = (j < 24) ? -xb[c + 24] : xb[c - 24];
        outv = xn * cs + rot * sn;
    } else {
        outv = xn;
    }
    const u16 bfv = f2bf(outv);
    if (hh < 16) {
        Qb[((size_t)(b * NH_ + hh) * S_ + s) * HD_ + c] = bfv;
    } else if (hh < 24) {
        Kb[((size_t)(b * NKV_ + (hh - 16)) * S_ + s) * HD_ + c] = bfv;
    } else {
        Vt[((size_t)((b * NKV_ + (hh - 24)) * HD_ + c)) * S_ + s] = bfv;   // [b][h][d][s]
    }
}

// ---------------- kernel 3: scores = Q K^T (+mask), batched over (b,h) ----------------
__global__ __launch_bounds__(256) void scores_bf16(
    const u16* __restrict__ Qb, const u16* __restrict__ Kb,
    const unsigned char* __restrict__ mask, float* __restrict__ attn)
{
    const int bh = blockIdx.z;
    const int b = bh >> 4, qh = bh & 15, kh = qh >> 1;   // GQA n_rep = 2
    const u16* A    = Qb + (size_t)bh * S_ * HD_;
    const u16* Bsrc = Kb + (size_t)(b * NKV_ + kh) * S_ * HD_;
    float* C = attn + (size_t)bh * S_ * S_;
    const unsigned char* mk = mask + (size_t)b * S_ * S_;

    __shared__ __align__(16) u16 As[2][64][40];
    __shared__ __align__(16) u16 Bs[2][64][40];
    const int m0 = blockIdx.y * 64, n0 = blockIdx.x * 64;
    const int t = threadIdx.x, lane = t & 31, wid = t >> 5;
    const int wrow = (wid & 3) * 16, wcol = (wid >> 2) * 32;
    const int lr = lane & 15, hi = lane >> 4;
    const int row = t >> 2, seg = (t & 3) * 8;
    v8f acc0 = {}, acc1 = {};

    auto issue = [&](int k0, int buf) {
        async_b128(lds_off(&As[buf][row][seg]), &A   [(size_t)(m0 + row) * HD_ + k0 + seg]);
        async_b128(lds_off(&Bs[buf][row][seg]), &Bsrc[(size_t)(n0 + row) * HD_ + k0 + seg]);
    };

    issue(0, 0);
    for (int i = 0; i < 3; ++i) {         // K = 96 = 3 x 32
        const int buf = i & 1;
        if (i + 1 < 3) { issue((i + 1) << 5, buf ^ 1); WAIT_ASYNC(2); }
        else           { WAIT_ASYNC(0); }
        __syncthreads();

        Frag16 a, b0, b1;
        const u16* ap = &As[buf][wrow + lr][0];
        a.h[0] = *(const u16x8*)&ap[hi * 8];
        a.h[1] = *(const u16x8*)&ap[16 + hi * 8];
        const u16* bp0 = &Bs[buf][wcol + lr][hi * 8];
        b0.h[0] = *(const u16x8*)&bp0[0];
        b0.h[1] = *(const u16x8*)&bp0[16];
        const u16* bp1 = &Bs[buf][wcol + 16 + lr][hi * 8];
        b1.h[0] = *(const u16x8*)&bp1[0];
        b1.h[1] = *(const u16x8*)&bp1[16];

        acc0 = __builtin_amdgcn_wmma_f32_16x16x32_bf16(false, a.bf, false, b0.bf, (short)0, acc0, false, false);
        acc1 = __builtin_amdgcn_wmma_f32_16x16x32_bf16(false, a.bf, false, b1.bf, (short)0, acc1, false, false);
        __syncthreads();
    }
#pragma unroll
    for (int r = 0; r < 8; ++r) {
        const int m = m0 + wrow + r + hi * 8;
        const int n = n0 + wcol + lr;
        float* cp = &C[(size_t)m * S_ + n];
        cp[0]  = mk[(size_t)m * S_ + n]      ? acc0[r] : NEGINF_;
        cp[16] = mk[(size_t)m * S_ + n + 16] ? acc1[r] : NEGINF_;
    }
}

// ---------------- kernel 4: in-place row softmax (fp32) ----------------
__global__ __launch_bounds__(256) void softmax_rows(float* __restrict__ attn)
{
    __shared__ float sred[256];
    const int t = threadIdx.x;
    float* p = attn + (size_t)blockIdx.x * S_;
    float v[8];
    float mx = NEGINF_;
#pragma unroll
    for (int i = 0; i < 8; ++i) { v[i] = p[t + i * 256]; mx = fmaxf(mx, v[i]); }
    sred[t] = mx; __syncthreads();
    for (int s2 = 128; s2 > 0; s2 >>= 1) { if (t < s2) sred[t] = fmaxf(sred[t], sred[t + s2]); __syncthreads(); }
    mx = sred[0]; __syncthreads();
    float sum = 0.f;
#pragma unroll
    for (int i = 0; i < 8; ++i) { v[i] = expf(v[i] - mx); sum += v[i]; }
    sred[t] = sum; __syncthreads();
    for (int s2 = 128; s2 > 0; s2 >>= 1) { if (t < s2) sred[t] += sred[t + s2]; __syncthreads(); }
    const float inv = 1.0f / sred[0];
#pragma unroll
    for (int i = 0; i < 8; ++i) p[t + i * 256] = v[i] * inv;
}

// ---------------- kernel 5: ctx = P @ V, batched over (b,h) ----------------
// N = HD_ = 96 (full), M tile 64, K loop over 2048 keys, step 32.
// 8 waves: 4 M-tiles x 2 N-halves (48 = 3x16), each wave 3 accumulators.
// Probs tile async-loaded as fp32 (converted to bf16 at fragment build),
// V tile async-loaded as bf16.  5 async instrs per wave per tile (uniform).
__global__ __launch_bounds__(256) void pv_bf16(
    const float* __restrict__ attn, const u16* __restrict__ Vt,
    u16* __restrict__ ctx)
{
    const int bh = blockIdx.z;
    const int b = bh >> 4, qh = bh & 15, vh = qh >> 1;
    const float* P = attn + (size_t)bh * S_ * S_;
    const u16*   V = Vt + (size_t)(b * NKV_ + vh) * HD_ * S_;   // [d][s]
    const int m0 = blockIdx.x * 64;

    __shared__ __align__(16) float As32[2][64][36];
    __shared__ __align__(16) u16   Bs[2][96][40];
    const int t = threadIdx.x, lane = t & 31, wid = t >> 5;
    const int wrow = (wid & 3) * 16, wcol = (wid >> 2) * 48;
    const int lr = lane & 15, hi = lane >> 4;
    const int arow = t >> 2, aseg = (t & 3) * 8;   // 8 fp32 per thread
    v8f acc[3] = {{}, {}, {}};

    auto issue = [&](int k0, int buf) {
        const float* gp = &P[(size_t)(m0 + arow) * S_ + k0 + aseg];
        async_b128(lds_off(&As32[buf][arow][aseg]),     gp);
        async_b128(lds_off(&As32[buf][arow][aseg + 4]), gp + 4);
#pragma unroll
        for (int j = 0; j < 3; ++j) {                  // 768 b64 segs / 256 threads
            const int sidx = t + j * 256;
            const int n = sidx >> 3, s4 = (sidx & 7) * 4;
            async_b64(lds_off(&Bs[buf][n][s4]), &V[(size_t)n * S_ + k0 + s4]);
        }
    };

    const int nt = S_ >> 5;   // 64
    issue(0, 0);
    for (int i = 0; i < nt; ++i) {
        const int buf = i & 1;
        if (i + 1 < nt) { issue((i + 1) << 5, buf ^ 1); WAIT_ASYNC(5); }
        else            { WAIT_ASYNC(0); }
        __syncthreads();

        Frag16 a;
        const float* ap = &As32[buf][wrow + lr][0];
#pragma unroll
        for (int ii = 0; ii < 8; ++ii) {
            a.u[ii]     = f2bf(ap[hi * 8 + ii]);
            a.u[ii + 8] = f2bf(ap[16 + hi * 8 + ii]);
        }
#pragma unroll
        for (int f = 0; f < 3; ++f) {
            Frag16 bfg;
            const u16* bp = &Bs[buf][wcol + f * 16 + lr][hi * 8];
            bfg.h[0] = *(const u16x8*)&bp[0];
            bfg.h[1] = *(const u16x8*)&bp[16];
            acc[f] = __builtin_amdgcn_wmma_f32_16x16x32_bf16(false, a.bf, false, bfg.bf, (short)0, acc[f], false, false);
        }
        __syncthreads();
    }
#pragma unroll
    for (int r = 0; r < 8; ++r) {
        const int m = m0 + wrow + r + hi * 8;
        const size_t base = ((size_t)(b * S_ + m)) * HID_ + qh * HD_ + wcol + lr;
#pragma unroll
        for (int f = 0; f < 3; ++f) ctx[base + f * 16] = f2bf(acc[f][r]);
    }
}

// ---------------- host launcher ----------------
extern "C" void kernel_launch(void* const* d_in, const int* in_sizes, int n_in,
                              void* d_out, int out_size, void* d_ws, size_t ws_size,
                              hipStream_t stream) {
    (void)in_sizes; (void)n_in; (void)out_size; (void)ws_size;
    const float* hidden = (const float*)d_in[0];
    const float* cosp   = (const float*)d_in[1];
    const float* sinp   = (const float*)d_in[2];
    const unsigned char* mask = (const unsigned char*)d_in[3];
    const float* Wq = (const float*)d_in[4];
    const float* Wk = (const float*)d_in[5];
    const float* Wv = (const float*)d_in[6];
    const float* Wo = (const float*)d_in[7];
    const float* qw = (const float*)d_in[8];
    const float* kw = (const float*)d_in[9];

    float* out   = (float*)d_out;                               // (B,S,HID)
    float* attnw = out + (size_t)B_ * S_ * HID_;                // (B,NH,S,S)

    // carve workspace (256B aligned regions)
    char* w = (char*)d_ws;
    auto carve = [&](size_t bytes) { void* p = (void*)w; w += (bytes + 255) & ~(size_t)255; return p; };
    const size_t M  = (size_t)B_ * S_;                          // 8192
    u16*   hidden_bf = (u16*)  carve(M * HID_ * 2);
    u16*   wqkv_bf   = (u16*)  carve((size_t)QKVN_ * HID_ * 2);
    u16*   wo_bf     = (u16*)  carve((size_t)HID_ * HID_ * 2);
    float* qkv       = (float*)carve(M * QKVN_ * 4);
    u16*   Qb        = (u16*)  carve((size_t)B_ * NH_  * S_ * HD_ * 2);
    u16*   Kb        = (u16*)  carve((size_t)B_ * NKV_ * S_ * HD_ * 2);
    u16*   Vtb       = (u16*)  carve((size_t)B_ * NKV_ * HD_ * S_ * 2);
    u16*   ctxb      = (u16*)  carve(M * HID_ * 2);

    // 0) convert fp32 -> bf16 (Wq||Wk||Wv concatenated row-wise)
    {
        const int nh = (int)(M * HID_);
        cvt_f32_bf16<<<(nh + 255) / 256, 256, 0, stream>>>(hidden, hidden_bf, nh);
        const int nq = NH_ * HD_ * HID_, nk = NKV_ * HD_ * HID_;
        cvt_f32_bf16<<<(nq + 255) / 256, 256, 0, stream>>>(Wq, wqkv_bf, nq);
        cvt_f32_bf16<<<(nk + 255) / 256, 256, 0, stream>>>(Wk, wqkv_bf + (size_t)nq, nk);
        cvt_f32_bf16<<<(nk + 255) / 256, 256, 0, stream>>>(Wv, wqkv_bf + (size_t)nq + nk, nk);
        const int no = HID_ * HID_;
        cvt_f32_bf16<<<(no + 255) / 256, 256, 0, stream>>>(Wo, wo_bf, no);
    }

    // 1) fused QKV projection: (8192x1536) @ (3072x1536)^T -> fp32
    gemm_nt_bf16<<<dim3(QKVN_ / 64, M / 64), 256, 0, stream>>>(
        hidden_bf, wqkv_bf, qkv, (int)M, QKVN_, HID_);

    // 2) RMSNorm + mRoPE -> bf16 Q/K, transposed bf16 V
    normrope_kernel<<<(int)(M * 32), 96, 0, stream>>>(qkv, cosp, sinp, qw, kw, Qb, Kb, Vtb);

    // 3) scores = Q K^T + mask, into attn_w output region
    scores_bf16<<<dim3(S_ / 64, S_ / 64, B_ * NH_), 256, 0, stream>>>(Qb, Kb, mask, attnw);

    // 4) in-place softmax over each row of attn_w
    softmax_rows<<<B_ * NH_ * S_, 256, 0, stream>>>(attnw);

    // 5) ctx = P @ V (bf16 context in (B,S,NH*HD) layout)
    pv_bf16<<<dim3(S_ / 64, 1, B_ * NH_), 256, 0, stream>>>(attnw, Vtb, ctxb);

    // 6) out = ctx @ Wo^T
    gemm_nt_bf16<<<dim3(HID_ / 64, M / 64), 256, 0, stream>>>(
        ctxb, wo_bf, out, (int)M, HID_, HID_);
}
